// MultiHeadAttentionBlock_14336600834323
// MI455X (gfx1250) — matrix operand order repaired and checked by reading, not compiled, not run on hardware
//
#include <hip/hip_runtime.h>
#include <hip/hip_bf16.h>
#include <cstdint>

// Problem constants (from reference): B=2, S=2048, D=1024, H=16, DK=64
#define BB  2
#define SS  2048
#define DD  1024
#define HH  16
#define DKK 64
#define MM  (BB * SS)   // 4096 rows for the projection GEMMs

typedef __attribute__((ext_vector_type(16))) __bf16 v16bf;
typedef __attribute__((ext_vector_type(8)))  float  v8f;
typedef __attribute__((ext_vector_type(4)))  unsigned int u32x4;
typedef __attribute__((ext_vector_type(8)))  int   i32x8;
typedef __attribute__((ext_vector_type(4)))  int   i32x4;

union Frag16 {      // one WMMA 16-bf16 operand = 8 VGPRs = 32 bytes
    v16bf v;
    uint4 u[2];
    __bf16 h[16];
};
union BF16x8 {      // 8 packed bf16 for 16B vector stores
    uint4 u4;
    __bf16 h[8];
};

// Build a v16bf from two 8-float chunks (fp32 -> bf16 convert in regs).
__device__ inline v16bf cvt_ab(float4 a0, float4 a1, float4 b0, float4 b1) {
    Frag16 f;
    f.h[0]  = (__bf16)a0.x; f.h[1]  = (__bf16)a0.y; f.h[2]  = (__bf16)a0.z; f.h[3]  = (__bf16)a0.w;
    f.h[4]  = (__bf16)a1.x; f.h[5]  = (__bf16)a1.y; f.h[6]  = (__bf16)a1.z; f.h[7]  = (__bf16)a1.w;
    f.h[8]  = (__bf16)b0.x; f.h[9]  = (__bf16)b0.y; f.h[10] = (__bf16)b0.z; f.h[11] = (__bf16)b0.w;
    f.h[12] = (__bf16)b1.x; f.h[13] = (__bf16)b1.y; f.h[14] = (__bf16)b1.z; f.h[15] = (__bf16)b1.w;
    return f.v;
}

// ---------------------------------------------------------------------------
// TDM: issue a 2D bf16 tile DMA (global -> LDS).  D# packed per CDNA5 ISA
// (group0: count/lds_addr/global_addr/type=2; group1: data_size=2B, dims,
//  tiles, dim0 stride).  Groups 2/3 zero (<=2D tensor).
// ---------------------------------------------------------------------------
__device__ inline void tdm_load_2d(uint64_t gaddr, uint32_t lds_addr,
                                   uint32_t dim0, uint32_t dim1,
                                   uint32_t tile0, uint32_t tile1,
                                   uint32_t stride0)
{
    u32x4 g0;
    g0[0] = 1u;                                            // count=1 (valid user D#)
    g0[1] = lds_addr;                                      // bits 63:32 lds_addr
    g0[2] = (uint32_t)(gaddr & 0xFFFFFFFFull);             // global_addr[31:0]
    g0[3] = (uint32_t)((gaddr >> 32) & 0x01FFFFFFull)      // global_addr[56:32]
          | (2u << 30);                                    // type=2 ("image")

    i32x8 g1;
    g1[0] = (int)(1u << 16);                               // data_size=1 -> 2 bytes
    g1[1] = (int)((dim0 & 0xFFFFu) << 16);                 // tensor_dim0[15:0]
    g1[2] = (int)((dim0 >> 16) | ((dim1 & 0xFFFFu) << 16));// dim0[31:16], dim1[15:0]
    g1[3] = (int)((dim1 >> 16) | (tile0 << 16));           // dim1[31:16], tile_dim0
    g1[4] = (int)(tile1 & 0xFFFFu);                        // tile_dim1 (tile_dim2=0)
    g1[5] = (int)stride0;                                  // tensor_dim0_stride[31:0]
    g1[6] = 0;                                             // stride0[47:32], stride1[15:0]
    g1[7] = 0;                                             // stride1[47:16]

    i32x4 z4 = {0, 0, 0, 0};
#if __has_include(<hip/amd_detail/amd_gfx1250_TDM.h>)
    i32x8 z8 = {0, 0, 0, 0, 0, 0, 0, 0};
    __builtin_amdgcn_tensor_load_to_lds(g0, g1, z4, z4, z8, 0);
#else
    __builtin_amdgcn_tensor_load_to_lds(g0, g1, z4, z4, 0);
#endif
}

// ---------------------------------------------------------------------------
// GEMM: C[M,N] = A[M,K] * W[N,K]^T + bias[N]
// MODE 0: store bf16 row-major [M,N]          (Q/K projections)
// MODE 1: store bf16 transposed [B][N][S]     (V projection -> V^T for PV WMMA)
// MODE 2: store f32 row-major [M,N]           (output projection)
// Block = 256 threads = 8 waves (4 along M x 2 along N); wave tile 32x32.
// ---------------------------------------------------------------------------
template <int MODE>
__global__ __launch_bounds__(256) void gemm_bias_kernel(
    const float* __restrict__ A, const float* __restrict__ W,
    const float* __restrict__ bias, void* __restrict__ out,
    int M, int N, int K)
{
    const int lane = threadIdx.x & 31;
    const int wave = threadIdx.x >> 5;
    const int hi   = lane >> 4;       // half-wave select
    const int ln   = lane & 15;
    const int mb   = blockIdx.x * 128 + (wave >> 1) * 32;
    const int nb   = blockIdx.y * 64  + (wave & 1)  * 32;

    v8f acc[2][2];
    const v8f vzero = {0.f, 0.f, 0.f, 0.f, 0.f, 0.f, 0.f, 0.f};
#pragma unroll
    for (int mi = 0; mi < 2; ++mi)
#pragma unroll
        for (int ni = 0; ni < 2; ++ni) acc[mi][ni] = vzero;

    for (int k0 = 0; k0 < K; k0 += 32) {
        // Prefetch next K-tile into cache (global_prefetch_b8 path).
        if (k0 + 64 < K) {
            __builtin_prefetch(A + (size_t)(mb + ln) * K + k0 + 64, 0, 3);
            __builtin_prefetch(W + (size_t)(nb + ln) * K + k0 + 64, 0, 3);
        }
        v16bf af[2], bf[2];
#pragma unroll
        for (int i = 0; i < 2; ++i) {   // A frags: M = ln, K chunks hi*8 and hi*8+16
            const float* p = A + (size_t)(mb + i * 16 + ln) * K + k0 + hi * 8;
            float4 x0 = ((const float4*)p)[0];
            float4 x1 = ((const float4*)p)[1];
            float4 y0 = ((const float4*)(p + 16))[0];
            float4 y1 = ((const float4*)(p + 16))[1];
            af[i] = cvt_ab(x0, x1, y0, y1);
        }
#pragma unroll
        for (int i = 0; i < 2; ++i) {   // B frags: N = ln, K chunks hi*8 and hi*8+16
            const float* p = W + (size_t)(nb + i * 16 + ln) * K + k0 + hi * 8;
            float4 x0 = ((const float4*)p)[0];
            float4 x1 = ((const float4*)p)[1];
            float4 x2 = ((const float4*)(p + 16))[0];
            float4 x3 = ((const float4*)(p + 16))[1];
            bf[i] = cvt_ab(x0, x1, x2, x3);
        }
#pragma unroll
        for (int mi = 0; mi < 2; ++mi)
#pragma unroll
            for (int ni = 0; ni < 2; ++ni)
                acc[mi][ni] = __builtin_amdgcn_wmma_f32_16x16x32_bf16(
                    false, af[mi], false, bf[ni], (short)0, acc[mi][ni], false, false);
    }

#pragma unroll
    for (int mi = 0; mi < 2; ++mi) {
#pragma unroll
        for (int ni = 0; ni < 2; ++ni) {
            const int n  = nb + ni * 16 + ln;
            const float bv = bias[n];
            if (MODE == 1) {
                // Transposed bf16 store: rows r..r+7 are consecutive s -> one 16B store.
                const int m0   = mb + mi * 16 + hi * 8;
                const int bidx = m0 / SS;
                const int s    = m0 % SS;
                BF16x8 pk;
#pragma unroll
                for (int r = 0; r < 8; ++r) pk.h[r] = (__bf16)(acc[mi][ni][r] + bv);
                *(uint4*)((__bf16*)out + ((size_t)bidx * N + n) * SS + s) = pk.u4;
            } else {
#pragma unroll
                for (int r = 0; r < 8; ++r) {
                    const int m = mb + mi * 16 + hi * 8 + r;
                    const float v = acc[mi][ni][r] + bv;
                    if (MODE == 0) ((__bf16*)out)[(size_t)m * N + n] = (__bf16)v;
                    else           ((float*)out)[(size_t)m * N + n]  = v;
                }
            }
        }
    }
}

// ---------------------------------------------------------------------------
// Flash attention: grid = (B*H, S/128), block = 256 (8 waves x 16 queries).
// Qp/Kp: bf16 [B*S, D] row-major.  VpT: bf16 [B][D][S].  ctx: f32 [B*S, D].
// K/V tiles staged into LDS by the Tensor Data Mover (double-buffered),
// all 8 waves read B-fragments from LDS (8x less global traffic).
// ---------------------------------------------------------------------------
__global__ __launch_bounds__(256) void attn_kernel(
    const __bf16* __restrict__ Qp, const __bf16* __restrict__ Kp,
    const __bf16* __restrict__ VpT, const int* __restrict__ mask,
    float* __restrict__ ctx)
{
    __shared__ __bf16 ktile[2][32][64];   // [buf][key][d]   4KB each
    __shared__ __bf16 vtile[2][64][32];   // [buf][d][key]   4KB each
    __shared__ __bf16 plds[8][16 * 32];   // wave-private P-tile bounce (C->A transpose)

    const int lane = threadIdx.x & 31;
    const int wave = threadIdx.x >> 5;
    const int hi   = lane >> 4;
    const int ln   = lane & 15;
    const int b    = blockIdx.x >> 4;
    const int h    = blockIdx.x & 15;
    const int q0   = blockIdx.y * 128 + wave * 16;

    // Q fragments (16 queries x 64 dims) stay in registers for the whole kernel.
    v16bf qa[2];
#pragma unroll
    for (int i = 0; i < 2; ++i) {
        const __bf16* p = Qp + (size_t)(b * SS + q0 + ln) * DD + h * DKK + i * 32 + hi * 16;
        Frag16 f;
        f.u[0] = *(const uint4*)p;
        f.u[1] = *(const uint4*)(p + 8);
        qa[i] = f.v;
    }

    const v8f vzero = {0.f, 0.f, 0.f, 0.f, 0.f, 0.f, 0.f, 0.f};
    v8f o[4];
#pragma unroll
    for (int t = 0; t < 4; ++t) o[t] = vzero;
    float mrun[8], lrun[8], alpha[8];
#pragma unroll
    for (int r = 0; r < 8; ++r) { mrun[r] = -1e30f; lrun[r] = 0.f; }

    // Prologue: wave 0 DMAs the first K/V tiles into buffer 0.
    if (wave == 0) {
        tdm_load_2d((uint64_t)(uintptr_t)(Kp + (size_t)(b * SS) * DD + h * DKK),
                    (uint32_t)(uintptr_t)&ktile[0][0][0], 64, 32, 64, 32, DD);
        tdm_load_2d((uint64_t)(uintptr_t)(VpT + (size_t)(b * DD + h * DKK) * SS),
                    (uint32_t)(uintptr_t)&vtile[0][0][0], 32, 64, 32, 64, SS);
    }

    for (int j0 = 0; j0 < SS; j0 += 32) {
        const int cur = (j0 >> 5) & 1;
        const int nxt = cur ^ 1;

        // Issue DMA for the next tile pair, then wait for the current pair.
        if (wave == 0) {
            if (j0 + 32 < SS) {
                tdm_load_2d((uint64_t)(uintptr_t)(Kp + (size_t)(b * SS + j0 + 32) * DD + h * DKK),
                            (uint32_t)(uintptr_t)&ktile[nxt][0][0], 64, 32, 64, 32, DD);
                tdm_load_2d((uint64_t)(uintptr_t)(VpT + (size_t)(b * DD + h * DKK) * SS + j0 + 32),
                            (uint32_t)(uintptr_t)&vtile[nxt][0][0], 32, 64, 32, 64, SS);
                __builtin_amdgcn_s_wait_tensorcnt(2);   // in-order: current pair done
            } else {
                __builtin_amdgcn_s_wait_tensorcnt(0);
            }
        }
        __syncthreads();   // publish LDS tiles to all waves

        // ---- scores: S = Q (16x64) * K^T (64x32), two 16x16 C tiles ----
        v8f c[2];
#pragma unroll
        for (int t = 0; t < 2; ++t) {
            v16bf kb0, kb1;
            {
                const __bf16* p = &ktile[cur][t * 16 + ln][hi * 16];
                Frag16 f; f.u[0] = *(const uint4*)p; f.u[1] = *(const uint4*)(p + 8); kb0 = f.v;
            }
            {
                const __bf16* p = &ktile[cur][t * 16 + ln][32 + hi * 16];
                Frag16 f; f.u[0] = *(const uint4*)p; f.u[1] = *(const uint4*)(p + 8); kb1 = f.v;
            }
            v8f ct = __builtin_amdgcn_wmma_f32_16x16x32_bf16(false, qa[0], false, kb0,
                                                             (short)0, vzero, false, false);
            c[t] = __builtin_amdgcn_wmma_f32_16x16x32_bf16(false, qa[1], false, kb1,
                                                           (short)0, ct, false, false);
        }

        // ---- scale + mask (reference: /sqrt(DK), where(mask==0, -1e9)) ----
#pragma unroll
        for (int t = 0; t < 2; ++t)
#pragma unroll
            for (int r = 0; r < 8; ++r) {
                float v = c[t][r] * 0.125f;
                const int qq  = q0 + r + hi * 8;
                const int key = j0 + t * 16 + ln;
                if (mask[((size_t)b * SS + qq) * SS + key] == 0) v = -1e9f;
                c[t][r] = v;
            }

        // ---- online softmax: row reductions across the 16 lanes of a half ----
#pragma unroll
        for (int r = 0; r < 8; ++r) {
            float rm = fmaxf(c[0][r], c[1][r]);
            rm = fmaxf(rm, __shfl_xor(rm, 1, 32));
            rm = fmaxf(rm, __shfl_xor(rm, 2, 32));
            rm = fmaxf(rm, __shfl_xor(rm, 4, 32));
            rm = fmaxf(rm, __shfl_xor(rm, 8, 32));
            const float mnew = fmaxf(mrun[r], rm);
            alpha[r] = __expf(mrun[r] - mnew);
            mrun[r]  = mnew;
            const float p0 = __expf(c[0][r] - mnew);
            const float p1 = __expf(c[1][r] - mnew);
            c[0][r] = p0; c[1][r] = p1;
            float rs = p0 + p1;
            rs += __shfl_xor(rs, 1, 32);
            rs += __shfl_xor(rs, 2, 32);
            rs += __shfl_xor(rs, 4, 32);
            rs += __shfl_xor(rs, 8, 32);
            lrun[r] = lrun[r] * alpha[r] + rs;
        }

        // ---- P (C-layout f32) -> LDS bf16 -> A-fragment layout ----
        __bf16* pl = plds[wave];
#pragma unroll
        for (int t = 0; t < 2; ++t)
#pragma unroll
            for (int r = 0; r < 8; ++r)
                pl[(r + hi * 8) * 32 + t * 16 + ln] = (__bf16)c[t][r];

#pragma unroll
        for (int t = 0; t < 4; ++t)
#pragma unroll
            for (int r = 0; r < 8; ++r) o[t][r] *= alpha[r];

        Frag16 pf;
        const __bf16* pp = pl + ln * 32 + hi * 16;
        pf.u[0] = *(const uint4*)pp;
        pf.u[1] = *(const uint4*)(pp + 8);
        const v16bf pa = pf.v;

        // ---- O += P (16x32) * V (32x64); V^T tile in LDS gives contiguous B frags ----
#pragma unroll
        for (int t = 0; t < 4; ++t) {
            const __bf16* p = &vtile[cur][t * 16 + ln][hi * 16];
            Frag16 f; f.u[0] = *(const uint4*)p; f.u[1] = *(const uint4*)(p + 8);
            o[t] = __builtin_amdgcn_wmma_f32_16x16x32_bf16(false, pa, false, f.v,
                                                           (short)0, o[t], false, false);
        }

        __syncthreads();   // everyone done with buffer `cur` before it is refilled
    }

    // ---- normalize and write context [B*S, D] (f32) ----
    float invl[8];
#pragma unroll
    for (int r = 0; r < 8; ++r) invl[r] = 1.0f / lrun[r];
#pragma unroll
    for (int t = 0; t < 4; ++t)
#pragma unroll
        for (int r = 0; r < 8; ++r) {
            const int qq = q0 + r + hi * 8;
            const int d  = h * DKK + t * 16 + ln;
            ctx[(size_t)(b * SS + qq) * DD + d] = o[t][r] * invl[r];
        }
}

// ---------------------------------------------------------------------------
extern "C" void kernel_launch(void* const* d_in, const int* in_sizes, int n_in,
                              void* d_out, int out_size, void* d_ws, size_t ws_size,
                              hipStream_t stream) {
    (void)in_sizes; (void)n_in; (void)out_size; (void)ws_size;
    const float* qin  = (const float*)d_in[0];
    const float* kin  = (const float*)d_in[1];
    const float* vin  = (const float*)d_in[2];
    const int*   mask = (const int*)d_in[3];
    const float* wq = (const float*)d_in[4];
    const float* bq = (const float*)d_in[5];
    const float* wk = (const float*)d_in[6];
    const float* bk = (const float*)d_in[7];
    const float* wv = (const float*)d_in[8];
    const float* bv = (const float*)d_in[9];
    const float* wo = (const float*)d_in[10];
    const float* bo = (const float*)d_in[11];
    float* out = (float*)d_out;

    char* ws = (char*)d_ws;
    const size_t MB = (size_t)1 << 20;
    __bf16* Qp  = (__bf16*)(ws + 0 * MB);    // 8 MB  bf16 [B*S, D]
    __bf16* Kp  = (__bf16*)(ws + 8 * MB);    // 8 MB  bf16 [B*S, D]
    __bf16* VpT = (__bf16*)(ws + 16 * MB);   // 8 MB  bf16 [B][D][S]
    float*  ctx = (float*)(ws + 24 * MB);    // 16 MB f32  [B*S, D]

    const dim3 gblk(256);
    const dim3 ggrid(MM / 128, DD / 64);     // 32 x 16

    gemm_bias_kernel<0><<<ggrid, gblk, 0, stream>>>(qin, wq, bq, Qp,  MM, DD, DD);
    gemm_bias_kernel<0><<<ggrid, gblk, 0, stream>>>(kin, wk, bk, Kp,  MM, DD, DD);
    gemm_bias_kernel<1><<<ggrid, gblk, 0, stream>>>(vin, wv, bv, VpT, MM, DD, DD);

    attn_kernel<<<dim3(BB * HH, SS / 128), gblk, 0, stream>>>(Qp, Kp, VpT, mask, ctx);

    gemm_bias_kernel<2><<<ggrid, gblk, 0, stream>>>(ctx, wo, bo, out, MM, DD, DD);
}
